// RefinedFusionAttention_1511828488315
// MI455X (gfx1250) — matrix operand order repaired and checked by reading, not compiled, not run on hardware
//
#include <hip/hip_runtime.h>
#include <hip/hip_bf16.h>

typedef __attribute__((ext_vector_type(16))) _Float16 v16h;
typedef __attribute__((ext_vector_type(8)))  _Float16 v8h;
typedef __attribute__((ext_vector_type(8)))  float    v8f;
typedef __attribute__((ext_vector_type(4)))  int      v4i;

#define BLQ   10          // b*l
#define LQ    1024        // seq len (32*32)
#define DIMQ  256
#define HEADSQ 8
#define HDQ   32
#define ROWSQ (BLQ*LQ)    // 10240

// Async global->LDS staging path (CDNA5): guarded so both toolchains compile.
#if defined(__HIP_DEVICE_COMPILE__) && \
    __has_builtin(__builtin_amdgcn_global_load_async_to_lds_b128) && \
    __has_builtin(__builtin_amdgcn_s_wait_asynccnt)
#define ATTN_ASYNC 1
typedef __attribute__((address_space(1))) v4i* gv4i_p;
typedef __attribute__((address_space(3))) v4i* lv4i_p;
#else
#define ATTN_ASYNC 0
#endif

// ---------------- fragment helpers (CDNA5 wave32 WMMA layouts) ----------------

__device__ __forceinline__ v16h cat16(v8h lo, v8h hi) {
    return __builtin_shufflevector(lo, hi, 0,1,2,3,4,5,6,7,8,9,10,11,12,13,14,15);
}

// A-matrix 16x32 f16: lane m=l%16 is row; halves j -> K = lh*16+j
__device__ __forceinline__ v16h load_a_frag(const _Float16* base, int ld, int lane) {
    int m = lane & 15, lh = lane >> 4;
    const _Float16* p = base + (size_t)m * ld + lh * 16;
    v8h lo = *(const v8h*)p;
    v8h hi = *(const v8h*)(p + 8);
    return cat16(lo, hi);
}

// B-matrix 32x16 f16 from [N][K] storage: lane n=l%16 is column; halves j -> K = kh*16+j
__device__ __forceinline__ v16h load_b_frag(const _Float16* base, int ld, int lane) {
    int n = lane & 15, kh = lane >> 4;
    const _Float16* p = base + (size_t)n * ld + kh * 16;
    v8h lo = *(const v8h*)p;
    v8h hi = *(const v8h*)(p + 8);
    return cat16(lo, hi);
}

__device__ __forceinline__ v8f wmma16(v16h a, v16h b, v8f c) {
    return __builtin_amdgcn_wmma_f32_16x16x32_f16(false, a, false, b, (short)0, c, false, false);
}

// ---------------- conversion kernels ----------------

__global__ void cvt_h_kernel(const float* __restrict__ s, _Float16* __restrict__ d, int n) {
    for (int i = blockIdx.x * blockDim.x + threadIdx.x; i < n; i += gridDim.x * blockDim.x)
        d[i] = (_Float16)s[i];
}

// src[K][N] (f32) -> dst[N][K] (f16)
__global__ void cvt_t_kernel(const float* __restrict__ s, _Float16* __restrict__ d, int K, int N) {
    int n_ = K * N;
    for (int i = blockIdx.x * blockDim.x + threadIdx.x; i < n_; i += gridDim.x * blockDim.x) {
        int k = i / N, n = i % N;
        d[(size_t)n * K + k] = (_Float16)s[i];
    }
}

// ---------------- positional bias MLP: posw[i][j][h] (half) ----------------

__global__ void pos_bias_kernel(const float* __restrict__ Wp1, const float* __restrict__ bp1,
                                const float* __restrict__ Wp2, const float* __restrict__ bp2,
                                _Float16* __restrict__ posw) {
    int p = blockIdx.x * blockDim.x + threadIdx.x;
    if (p >= LQ * LQ) return;
    int i = p >> 10, j = p & 1023;
    float dy = (float)((i >> 5) - (j >> 5));
    float dx = (float)((i & 31) - (j & 31));
    float dist = sqrtf(dy * dy + dx * dx);
    float oh[HEADSQ];
#pragma unroll
    for (int h = 0; h < HEADSQ; ++h) oh[h] = bp2[h];
    for (int t = 0; t < 32; ++t) {
        float hv = dy * Wp1[t] + dx * Wp1[32 + t] + dist * Wp1[64 + t] + bp1[t];
        hv = fmaxf(hv, 0.f);
#pragma unroll
        for (int h = 0; h < HEADSQ; ++h) oh[h] += hv * Wp2[t * 8 + h];
    }
#pragma unroll
    for (int h = 0; h < HEADSQ; ++h) posw[(size_t)p * 8 + h] = (_Float16)oh[h];
}

// ---------------- feature-similarity projections: s[row][8] (f32) ----------------

__global__ void s_kernel(const float* __restrict__ x, const float* __restrict__ Ws1,
                         const float* __restrict__ bs1, const float* __restrict__ Ws2,
                         const float* __restrict__ bs2, float* __restrict__ sbuf) {
    int row = blockIdx.x * blockDim.x + threadIdx.x;
    if (row >= ROWSQ) return;
    float hid[32];
#pragma unroll
    for (int t = 0; t < 32; ++t) hid[t] = bs1[t];
    const float* xr = x + (size_t)row * DIMQ;
    for (int k = 0; k < DIMQ; ++k) {
        float xv = xr[k];
        const float* wr = Ws1 + k * 32;
#pragma unroll
        for (int t = 0; t < 32; ++t) hid[t] += xv * wr[t];
    }
    float o[8];
#pragma unroll
    for (int h = 0; h < 8; ++h) o[h] = bs2[h];
    for (int t = 0; t < 32; ++t) {
        float hv = fmaxf(hid[t], 0.f);
#pragma unroll
        for (int h = 0; h < 8; ++h) o[h] += hv * Ws2[t * 8 + h];
    }
#pragma unroll
    for (int h = 0; h < 8; ++h) sbuf[(size_t)row * 8 + h] = o[h];
}

// ---------------- QKV projection GEMM (WMMA f16, f32 accum) ----------------

__global__ __launch_bounds__(256) void qkv_kernel(
        const _Float16* __restrict__ xh,
        const _Float16* __restrict__ Wqt, const _Float16* __restrict__ Wkt, const _Float16* __restrict__ Wvt,
        const float* __restrict__ bq, const float* __restrict__ bk, const float* __restrict__ bv,
        _Float16* __restrict__ qh, _Float16* __restrict__ kh, _Float16* __restrict__ vh) {
    const int MT = ROWSQ / 16, NT = DIMQ / 16;
    int wid = blockIdx.x * (blockDim.x >> 5) + (threadIdx.x >> 5);
    int lane = threadIdx.x & 31;
    int mat = wid / (MT * NT);
    if (mat >= 3) return;
    int rem = wid % (MT * NT);
    int mt = rem / NT, nt = rem % NT;
    const _Float16* W = (mat == 0) ? Wqt : (mat == 1) ? Wkt : Wvt;
    const float* bias = (mat == 0) ? bq : (mat == 1) ? bk : bv;
    _Float16* dst     = (mat == 0) ? qh : (mat == 1) ? kh : vh;
    int m = lane & 15, lh = lane >> 4;
    v8f c = {};
    for (int kk = 0; kk < DIMQ; kk += 32) {
        v16h a = load_a_frag(xh + (size_t)(mt * 16) * DIMQ + kk, DIMQ, lane);
        v16h b = load_b_frag(W  + (size_t)(nt * 16) * DIMQ + kk, DIMQ, lane);
        c = wmma16(a, b, c);
    }
    int col = nt * 16 + m;
    float bb = bias[col];
#pragma unroll
    for (int r = 0; r < 8; ++r) {
        int row = mt * 16 + lh * 8 + r;
        dst[(size_t)row * DIMQ + col] = (_Float16)(c[r] + bb);
    }
}

// ---------------- fused flash attention ----------------
// block = 8 waves, one (batch, head); each wave owns 16 query rows.
// K chunks async-staged into LDS (ASYNCcnt); V staged transposed; WMMA QK^T and PV.

__global__ __launch_bounds__(256) void attn_kernel(
        const _Float16* __restrict__ qh, const _Float16* __restrict__ khb,
        const _Float16* __restrict__ vhb, const _Float16* __restrict__ posw,
        const float* __restrict__ sbuf, _Float16* __restrict__ aoh) {
    __shared__ __align__(16) _Float16 kl[128 * 32];      // [key][d]
    __shared__ __align__(16) _Float16 vt[32 * 128];      // [d][key]  (transposed)
    __shared__ __align__(16) _Float16 pl[8][16 * 32];    // per-wave P staging

    int tid = threadIdx.x, lane = tid & 31, w = tid >> 5;
    int blk = blockIdx.x;
    int b  = blk >> 6;
    int rem = blk & 63;
    int h  = rem >> 3;
    int qt = rem & 7;
    int q0 = qt * 128 + w * 16;
    int m = lane & 15, lh = lane >> 4;
    const float scale = 0.17677669529663687f; // 32^-0.5

    // persistent Q fragment (K-dim = head_dim = 32, exactly one WMMA step)
    v16h qa = load_a_frag(qh + ((size_t)(b * LQ + q0)) * DIMQ + h * HDQ, DIMQ, lane);

    // this lane's 8 query rows' s-vectors (for fsim bias)
    float sq[8][8];
#pragma unroll
    for (int r = 0; r < 8; ++r) {
        const float* sp = sbuf + ((size_t)(b * LQ + q0 + lh * 8 + r)) * 8;
#pragma unroll
        for (int t = 0; t < 8; ++t) sq[r][t] = sp[t];
    }

    float mst[8], lst[8];
    v8f o0 = {}, o1 = {};
#pragma unroll
    for (int r = 0; r < 8; ++r) { mst[r] = -1e30f; lst[r] = 0.f; }

    for (int chunk = 0; chunk < 8; ++chunk) {
        int key0 = chunk * 128;
        __syncthreads();
        // stage K chunk: [key][d], contiguous 16B units
#if ATTN_ASYNC
        for (int e = tid; e < (128 * 32 / 8); e += 256) {
            int key = e >> 2, d0 = (e & 3) * 8;
            const _Float16* g = khb + ((size_t)(b * LQ + key0 + key)) * DIMQ + h * HDQ + d0;
            _Float16* l = kl + key * 32 + d0;
            __builtin_amdgcn_global_load_async_to_lds_b128((gv4i_p)g, (lv4i_p)l, 0, 0);
        }
#else
        for (int e = tid; e < (128 * 32 / 8); e += 256) {
            int key = e >> 2, d0 = (e & 3) * 8;
            *(v8h*)(kl + key * 32 + d0) =
                *(const v8h*)(khb + ((size_t)(b * LQ + key0 + key)) * DIMQ + h * HDQ + d0);
        }
#endif
        // stage V chunk transposed (register path; transpose can't use async copy)
        for (int e = tid; e < 128 * 32; e += 256) {
            int key = e >> 5, d = e & 31;
            vt[d * 128 + key] = vhb[((size_t)(b * LQ + key0 + key)) * DIMQ + h * HDQ + d];
        }
#if ATTN_ASYNC
        __builtin_amdgcn_s_wait_asynccnt(0);
#endif
        __syncthreads();

        for (int sub = 0; sub < 8; ++sub) {
            int k16 = sub * 16;
            // B fragment for K^T: b[j] = K[key=k16+n][d=kh*16+j]
            v16h kf;
            {
                const _Float16* p = kl + (k16 + m) * 32 + lh * 16;
                kf = cat16(*(const v8h*)p, *(const v8h*)(p + 8));
            }
            v8f zero = {};
            v8f c = wmma16(qa, kf, zero);

            // biases: scale, positional, feature-similarity
            int keyg = key0 + k16 + m;  // global key for this lane's column
            float sk[8];
            {
                const float* sp = sbuf + ((size_t)(b * LQ + keyg)) * 8;
#pragma unroll
                for (int t = 0; t < 8; ++t) sk[t] = sp[t];
            }
#pragma unroll
            for (int r = 0; r < 8; ++r) {
                int qrow = q0 + lh * 8 + r;
                float pw = (float)posw[((size_t)qrow * LQ + keyg) * 8 + h];
                float fs = 0.f;
#pragma unroll
                for (int t = 0; t < 8; ++t) fs += sq[r][t] * sk[t];
                c[r] = c[r] * scale + pw + fs;
            }

            // online softmax over the 16-key subtile (row reduce across 16-lane half)
#pragma unroll
            for (int r = 0; r < 8; ++r) {
                float v = c[r];
                float mx = v;
#pragma unroll
                for (int off = 1; off < 16; off <<= 1) mx = fmaxf(mx, __shfl_xor(mx, off, 32));
                float nm = fmaxf(mst[r], mx);
                float corr = __expf(mst[r] - nm);
                float pv = __expf(v - nm);
                float rs = pv;
#pragma unroll
                for (int off = 1; off < 16; off <<= 1) rs += __shfl_xor(rs, off, 32);
                lst[r] = lst[r] * corr + rs;
                mst[r] = nm;
                o0[r] *= corr; o1[r] *= corr;
                c[r] = pv;
            }

            // stash P (half) into wave-private LDS slab; every 2 subtiles run PV WMMA (K=32)
            _Float16* pw_ = pl[w];
#pragma unroll
            for (int r = 0; r < 8; ++r)
                pw_[(lh * 8 + r) * 32 + ((sub & 1) << 4) + m] = (_Float16)c[r];

            if (sub & 1) {
                int kbase = (sub - 1) * 16; // chunk-local base of the 32-key span
                v16h pa;
                {
                    const _Float16* p = pl[w] + m * 32 + lh * 16;
                    pa = cat16(*(const v8h*)p, *(const v8h*)(p + 8));
                }
                v16h v0f, v1f;
                {
                    const _Float16* p = vt + (size_t)m * 128 + kbase + lh * 16;
                    v0f = cat16(*(const v8h*)p, *(const v8h*)(p + 8));
                }
                {
                    const _Float16* p = vt + (size_t)(16 + m) * 128 + kbase + lh * 16;
                    v1f = cat16(*(const v8h*)p, *(const v8h*)(p + 8));
                }
                o0 = wmma16(pa, v0f, o0);
                o1 = wmma16(pa, v1f, o1);
            }
        }
    }

    // epilogue: normalize and store half attn output [row][h*32+d]
#pragma unroll
    for (int r = 0; r < 8; ++r) {
        float inv = 1.f / lst[r];
        int row = q0 + lh * 8 + r;
        _Float16* op = aoh + ((size_t)(b * LQ + row)) * DIMQ + h * HDQ;
        op[m]      = (_Float16)(o0[r] * inv);
        op[16 + m] = (_Float16)(o1[r] * inv);
    }
}

// ---------------- output projection GEMM ----------------

__global__ __launch_bounds__(256) void proj_kernel(
        const _Float16* __restrict__ aoh, const _Float16* __restrict__ Wot,
        const float* __restrict__ bo, float* __restrict__ o2f, _Float16* __restrict__ o2h) {
    const int MT = ROWSQ / 16, NT = DIMQ / 16;
    int wid = blockIdx.x * (blockDim.x >> 5) + (threadIdx.x >> 5);
    if (wid >= MT * NT) return;
    int lane = threadIdx.x & 31;
    int mt = wid / NT, nt = wid % NT;
    int m = lane & 15, lh = lane >> 4;
    v8f c = {};
    for (int kk = 0; kk < DIMQ; kk += 32) {
        v16h a = load_a_frag(aoh + (size_t)(mt * 16) * DIMQ + kk, DIMQ, lane);
        v16h b = load_b_frag(Wot + (size_t)(nt * 16) * DIMQ + kk, DIMQ, lane);
        c = wmma16(a, b, c);
    }
    int col = nt * 16 + m;
    float bb = bo[col];
#pragma unroll
    for (int r = 0; r < 8; ++r) {
        int row = mt * 16 + lh * 8 + r;
        float val = c[r] + bb;
        size_t idx = (size_t)row * DIMQ + col;
        o2f[idx] = val;
        o2h[idx] = (_Float16)val;
    }
}

// ---------------- gate GEMM (K=512: [x | out]) + sigmoid blend ----------------

__global__ __launch_bounds__(256) void gate_kernel(
        const _Float16* __restrict__ xh, const _Float16* __restrict__ o2h,
        const _Float16* __restrict__ Wgt, const float* __restrict__ bg,
        const float* __restrict__ xf, const float* __restrict__ o2f,
        float* __restrict__ outp) {
    const int MT = ROWSQ / 16, NT = DIMQ / 16;
    int wid = blockIdx.x * (blockDim.x >> 5) + (threadIdx.x >> 5);
    if (wid >= MT * NT) return;
    int lane = threadIdx.x & 31;
    int mt = wid / NT, nt = wid % NT;
    int m = lane & 15, lh = lane >> 4;
    v8f c = {};
    for (int kk = 0; kk < 512; kk += 32) {
        const _Float16* abase = (kk < 256)
            ? (xh  + (size_t)(mt * 16) * DIMQ + kk)
            : (o2h + (size_t)(mt * 16) * DIMQ + (kk - 256));
        v16h a = load_a_frag(abase, DIMQ, lane);
        v16h b = load_b_frag(Wgt + (size_t)(nt * 16) * 512 + kk, 512, lane);
        c = wmma16(a, b, c);
    }
    int col = nt * 16 + m;
    float bb = bg[col];
#pragma unroll
    for (int r = 0; r < 8; ++r) {
        int row = mt * 16 + lh * 8 + r;
        size_t idx = (size_t)row * DIMQ + col;
        float g = 1.f / (1.f + __expf(-(c[r] + bb)));
        outp[idx] = g * o2f[idx] + (1.f - g) * xf[idx];
    }
}

// ---------------- launch ----------------

extern "C" void kernel_launch(void* const* d_in, const int* in_sizes, int n_in,
                              void* d_out, int out_size, void* d_ws, size_t ws_size,
                              hipStream_t stream) {
    (void)in_sizes; (void)n_in; (void)out_size; (void)ws_size;
    const float* x   = (const float*)d_in[0];
    const float* Wq  = (const float*)d_in[1];
    const float* bq  = (const float*)d_in[2];
    const float* Wk  = (const float*)d_in[3];
    const float* bk  = (const float*)d_in[4];
    const float* Wv  = (const float*)d_in[5];
    const float* bv  = (const float*)d_in[6];
    const float* Wp1 = (const float*)d_in[7];
    const float* bp1 = (const float*)d_in[8];
    const float* Wp2 = (const float*)d_in[9];
    const float* bp2 = (const float*)d_in[10];
    const float* Ws1 = (const float*)d_in[11];
    const float* bs1 = (const float*)d_in[12];
    const float* Ws2 = (const float*)d_in[13];
    const float* bs2 = (const float*)d_in[14];
    const float* Wo  = (const float*)d_in[15];
    const float* bo  = (const float*)d_in[16];
    const float* Wg  = (const float*)d_in[17];
    const float* bg  = (const float*)d_in[18];

    char* wp = (char*)d_ws;
    auto carve = [&](size_t bytes) -> char* {
        char* p = wp;
        wp += (bytes + 255) & ~(size_t)255;
        return p;
    };
    const size_t actH = (size_t)ROWSQ * DIMQ * sizeof(_Float16); // 5.24 MB
    _Float16* xh  = (_Float16*)carve(actH);
    _Float16* qh  = (_Float16*)carve(actH);
    _Float16* kh  = (_Float16*)carve(actH);
    _Float16* vh  = (_Float16*)carve(actH);
    _Float16* aoh = (_Float16*)carve(actH);
    _Float16* o2h = (_Float16*)carve(actH);
    _Float16* Wqt = (_Float16*)carve((size_t)DIMQ * DIMQ * 2);
    _Float16* Wkt = (_Float16*)carve((size_t)DIMQ * DIMQ * 2);
    _Float16* Wvt = (_Float16*)carve((size_t)DIMQ * DIMQ * 2);
    _Float16* Wot = (_Float16*)carve((size_t)DIMQ * DIMQ * 2);
    _Float16* Wgt = (_Float16*)carve((size_t)DIMQ * 512 * 2);
    _Float16* posw = (_Float16*)carve((size_t)LQ * LQ * HEADSQ * 2); // 16.8 MB
    float* sbuf = (float*)carve((size_t)ROWSQ * 8 * 4);
    float* o2f  = (float*)carve((size_t)ROWSQ * DIMQ * 4);

    // 1. conversions / transposes
    cvt_h_kernel<<<2048, 256, 0, stream>>>(x, xh, ROWSQ * DIMQ);
    cvt_t_kernel<<<256, 256, 0, stream>>>(Wq, Wqt, DIMQ, DIMQ);
    cvt_t_kernel<<<256, 256, 0, stream>>>(Wk, Wkt, DIMQ, DIMQ);
    cvt_t_kernel<<<256, 256, 0, stream>>>(Wv, Wvt, DIMQ, DIMQ);
    cvt_t_kernel<<<256, 256, 0, stream>>>(Wo, Wot, DIMQ, DIMQ);
    cvt_t_kernel<<<512, 256, 0, stream>>>(Wg, Wgt, 512, DIMQ);

    // 2. positional bias (batch-independent, computed once)
    pos_bias_kernel<<<(LQ * LQ) / 256, 256, 0, stream>>>(Wp1, bp1, Wp2, bp2, posw);

    // 3. feature-similarity projections
    s_kernel<<<ROWSQ / 256, 256, 0, stream>>>(x, Ws1, bs1, Ws2, bs2, sbuf);

    // 4. QKV projections (WMMA)
    qkv_kernel<<<3 * (ROWSQ / 16) * (DIMQ / 16) / 8, 256, 0, stream>>>(
        xh, Wqt, Wkt, Wvt, bq, bk, bv, qh, kh, vh);

    // 5. fused flash attention (WMMA QK^T + online softmax + WMMA PV, async K staging)
    attn_kernel<<<BLQ * HEADSQ * (LQ / 128), 256, 0, stream>>>(qh, kh, vh, posw, sbuf, aoh);

    // 6. output projection (WMMA)
    proj_kernel<<<(ROWSQ / 16) * (DIMQ / 16) / 8, 256, 0, stream>>>(aoh, Wot, bo, o2f, o2h);

    // 7. gate GEMM (WMMA, K=512) + blend -> d_out
    gate_kernel<<<(ROWSQ / 16) * (DIMQ / 16) / 8, 256, 0, stream>>>(
        xh, o2h, Wgt, bg, x, o2f, (float*)d_out);
}
